// SchNet_48421461295178
// MI455X (gfx1250) — compile-verified, matrix operand order checked by reading, not run on hardware
//
#include <hip/hip_runtime.h>
#include <hip/hip_bf16.h>
#include <math.h>

// ---------------- problem constants ----------------
#define NATOMS 8192
#define APM    32
#define NMOL   256
#define FEAT   100
#define FPAD   128          // FEAT padded to 8x16
#define NG     25
#define NGPAD  32           // NG padded to one K=32 tile
#define KNN    28
#define NEDGE  (NATOMS*KNN) // 229376
#define NLAYER 4
#define CUTOFF 6.0f
#define ECHUNK 32           // edges processed per WMMA chunk
#define NCHUNK 28           // 896 edges / 32

typedef __attribute__((ext_vector_type(16))) _Float16 v16h;
typedef __attribute__((ext_vector_type(8)))  float    v8f;

// shifted softplus: softplus(x) - log(2)   (ssp(0) == 0, preserves zero padding)
__device__ __forceinline__ float sspf(float x) {
    float ax = fabsf(x);
    return fmaxf(x, 0.f) + __logf(1.f + __expf(-ax)) - 0.6931471805599453f;
}

// Load a 16x32 f16 fragment (A or B^T operand) from a row-major [rows][ld] buffer.
// CDNA5 16-bit A layout: lanes 0-15 hold row rT+lane, halves 0..7 = K 0..7,
// halves 8..15 = K 16..23; lanes 16-31 hold K 8..15 / 24..31.
__device__ __forceinline__ v16h load_frag(const _Float16* base, int ld, int rT, int kT) {
    int lane = threadIdx.x & 31;
    const _Float16* p = base + (size_t)(rT + (lane & 15)) * ld + kT + ((lane >> 4) << 3);
    v16h r;
#pragma unroll
    for (int i = 0; i < 8; i++) { r[i] = p[i]; r[8 + i] = p[16 + i]; }
    return r;
}

// Issue one 16-byte async global->LDS copy (per lane). LDS byte address is the
// low 32 bits of the generic pointer (ISA: LDS aperture addr truncates to [31:0]).
__device__ __forceinline__ void async_copy16(const _Float16* gsrc, const _Float16* ldst) {
    unsigned lds = (unsigned)(unsigned long long)(const void*)ldst;
    asm volatile("global_load_async_to_lds_b128 %0, %1, off"
                 :: "v"(lds), "v"(gsrc)
                 : "memory");
}

// ---------------- weight prep: f32 -> padded transposed f16 ----------------
__global__ void prep_kernel(const float* mw1, const float* mb1, const float* mw2, const float* mb2,
                            const float* l1w, const float* l2w, const float* l2b,
                            const float* lww, const float* lwb,
                            _Float16* W1p, _Float16* W2p, _Float16* L1p, _Float16* L2p, _Float16* LWp,
                            float* B1, float* B2, float* BL2, float* BLW) {
    int stride = gridDim.x * blockDim.x;
    int t0 = blockIdx.x * blockDim.x + threadIdx.x;
    // mlp_w1: (L,25,100) -> B^T padded [L][128][32]
    for (int idx = t0; idx < NLAYER * FPAD * NGPAD; idx += stride) {
        int l = idx >> 12, r = idx & 4095, n = r >> 5, k = r & 31;
        W1p[idx] = (n < FEAT && k < NG) ? (_Float16)mw1[(l * NG + k) * FEAT + n] : (_Float16)0.f;
    }
    // 100x100 weights -> B^T padded [L][128][128]
    for (int idx = t0; idx < NLAYER * FPAD * FPAD; idx += stride) {
        int l = idx >> 14, r = idx & 16383, n = r >> 7, k = r & 127;
        bool ok = (n < FEAT) && (k < FEAT);
        int s = (l * FEAT + k) * FEAT + n;
        W2p[idx] = ok ? (_Float16)mw2[s] : (_Float16)0.f;
        L1p[idx] = ok ? (_Float16)l1w[s] : (_Float16)0.f;
        L2p[idx] = ok ? (_Float16)l2w[s] : (_Float16)0.f;
        LWp[idx] = ok ? (_Float16)lww[s] : (_Float16)0.f;
    }
    // biases padded to 128
    for (int idx = t0; idx < NLAYER * FPAD; idx += stride) {
        int l = idx >> 7, n = idx & 127;
        bool ok = n < FEAT;
        int s = l * FEAT + n;
        B1[idx]  = ok ? mb1[s] : 0.f;
        B2[idx]  = ok ? mb2[s] : 0.f;
        BL2[idx] = ok ? l2b[s] : 0.f;
        BLW[idx] = ok ? lwb[s] : 0.f;
    }
}

// ---------------- embedding gather: h0 = emb[z], padded to 128 ----------------
__global__ void embed_kernel(const int* z, const float* emb, float* hbuf) {
    int idx = blockIdx.x * blockDim.x + threadIdx.x;
    if (idx >= NATOMS * FPAD) return;
    int a = idx >> 7, f = idx & 127;
    hbuf[idx] = (f < FEAT) ? emb[z[a] * FEAT + f] : 0.f;
}

// ---------------- molecule-local 28-NN graph ----------------
// One wave per molecule; thread i owns atom-row i. Edges at exactly CUTOFF get
// ccut==0 downstream, so in-molecule selection matches the reference numerics.
__global__ void graph_kernel(const float* pos, float* ew, int* srcLoc) {
    int m = blockIdx.x, i = threadIdx.x;   // 32 threads
    int a0 = m * APM;
    __shared__ float px[APM], py[APM], pz[APM];
    px[i] = pos[(a0 + i) * 3 + 0];
    py[i] = pos[(a0 + i) * 3 + 1];
    pz[i] = pos[(a0 + i) * 3 + 2];
    __syncthreads();
    float d[APM];
#pragma unroll
    for (int j = 0; j < APM; j++) {
        float dx = px[i] - px[j], dy = py[i] - py[j], dz = pz[i] - pz[j];
        float d2 = dx * dx + dy * dy + dz * dz;
        d[j] = (d2 > CUTOFF * CUTOFF) ? CUTOFF : __fsqrt_rn(d2);
    }
    d[i] = CUTOFF;                         // self masked
    for (int k = 0; k < KNN; k++) {        // selection: k-th smallest, ties by index
        float dmin = 1e30f; int jmin = 0;
#pragma unroll
        for (int j = 0; j < APM; j++) {
            if (d[j] < dmin) { dmin = d[j]; jmin = j; }
        }
        ew[(size_t)(a0 + i) * KNN + k] = dmin;
        srcLoc[(size_t)(a0 + i) * KNN + k] = jmin;   // local index within molecule
        d[jmin] = 1e30f;
    }
}

// ---------------- fused interaction block (one molecule per workgroup) ----------------
__global__ __launch_bounds__(128) void layer_kernel(
        float* hbuf, const float* ew, const int* srcLoc,
        const _Float16* W1p, const _Float16* W2p, const _Float16* L1p,
        const _Float16* L2p, const _Float16* LWp,
        const float* b1, const float* b2, const float* bl2, const float* blw) {
    __shared__ _Float16 hA[APM * FPAD];    // h (f16), later reused for agg (f16)
    __shared__ _Float16 x1A[APM * FPAD];   // x1 = h @ lin1
    __shared__ _Float16 tA[APM * FPAD];    // t1 / ssp(agg@lin2)
    __shared__ _Float16 wfA[ECHUNK * FPAD];// filter chunk
    __shared__ _Float16 eaA[ECHUNK * NGPAD];
    __shared__ _Float16 sW1[FPAD * NGPAD]; // mlp_w1 (B^T), reused 28x -> LDS resident
    __shared__ _Float16 sW2[FPAD * FPAD];  // mlp_w2 (B^T), reused 28x -> LDS resident
    __shared__ float agg[APM * FPAD];
    __shared__ float dS[ECHUNK], ccS[ECHUNK];
    __shared__ int   srcS[ECHUNK], tgtS[ECHUNK];

    const int tid = threadIdx.x, lane = tid & 31, wid = tid >> 5;
    const int a0 = blockIdx.x * APM;

    // async-stage the filter MLP weights into LDS (ASYNCcnt path), overlapped
    // with the synchronous f32->f16 staging of h below.
#pragma unroll
    for (int i = 0; i < (FPAD * NGPAD) / (8 * 128); i++)
        async_copy16(W1p + (tid + i * 128) * 8, sW1 + (tid + i * 128) * 8);
#pragma unroll
    for (int i = 0; i < (FPAD * FPAD) / (8 * 128); i++)
        async_copy16(W2p + (tid + i * 128) * 8, sW2 + (tid + i * 128) * 8);

    // warm late-use weights into cache (global_prefetch_b8)
    __builtin_prefetch(L2p + tid * 128, 0, 1);
    __builtin_prefetch(LWp + tid * 128, 0, 1);

    // stage h -> f16 LDS; zero agg
    for (int idx = tid; idx < APM * FPAD; idx += 128) {
        hA[idx] = (_Float16)hbuf[(size_t)(a0 + (idx >> 7)) * FPAD + (idx & 127)];
        agg[idx] = 0.f;
    }
    asm volatile("s_wait_asynccnt 0" ::: "memory");
    __syncthreads();

    // ---- x1 = h @ lin1 : 2x8 tiles, K=4x32 ----
    for (int t = wid; t < 16; t += 4) {
        int mT = (t >> 3) << 4, nT = (t & 7) << 4;
        v8f c = {};
#pragma unroll
        for (int kk = 0; kk < 4; kk++) {
            v16h a = load_frag(hA, FPAD, mT, kk * 32);
            v16h b = load_frag(L1p, FPAD, nT, kk * 32);
            c = __builtin_amdgcn_wmma_f32_16x16x32_f16(false, a, false, b, (short)0, c, false, false);
        }
        int n = nT + (lane & 15), m0 = mT + ((lane >> 4) << 3);
#pragma unroll
        for (int v = 0; v < 8; v++) x1A[(m0 + v) * FPAD + n] = (_Float16)c[v];
    }
    __syncthreads();

    // ---- edge loop: 28 chunks of 32 edges ----
    for (int ch = 0; ch < NCHUNK; ch++) {
        const size_t eBase = (size_t)a0 * KNN + ch * ECHUNK;
        if (tid < ECHUNK) {
            float d = ew[eBase + tid];
            dS[tid] = d;
            ccS[tid] = 0.5f * (__cosf(d * (3.14159265358979f / CUTOFF)) + 1.f);
            srcS[tid] = srcLoc[eBase + tid];
            tgtS[tid] = (ch * ECHUNK + tid) / KNN;
        }
        __syncthreads();
        // gaussian smearing: ea[e][g] = exp(-8*(d - 0.25 g)^2), pad g>=25 -> 0
        for (int q = 0; q < 8; q++) {
            int idx = tid + q * 128;
            int e = idx >> 5, g = idx & 31;
            float val = 0.f;
            if (g < NG) { float x = dS[e] - 0.25f * (float)g; val = __expf(-8.f * x * x); }
            eaA[idx] = (_Float16)val;
        }
        __syncthreads();
        // t1 = ssp(ea @ mlp_w1 + b1)
        for (int t = wid; t < 16; t += 4) {
            int mT = (t >> 3) << 4, nT = (t & 7) << 4;
            v16h a = load_frag(eaA, NGPAD, mT, 0);
            v16h b = load_frag(sW1, NGPAD, nT, 0);
            v8f c = {};
            c = __builtin_amdgcn_wmma_f32_16x16x32_f16(false, a, false, b, (short)0, c, false, false);
            int n = nT + (lane & 15), m0 = mT + ((lane >> 4) << 3);
            float bb = b1[n];
#pragma unroll
            for (int v = 0; v < 8; v++) tA[(m0 + v) * FPAD + n] = (_Float16)sspf(c[v] + bb);
        }
        __syncthreads();
        // Wf = (t1 @ mlp_w2 + b2) * ccut
        for (int t = wid; t < 16; t += 4) {
            int mT = (t >> 3) << 4, nT = (t & 7) << 4;
            v8f c = {};
#pragma unroll
            for (int kk = 0; kk < 4; kk++) {
                v16h a = load_frag(tA, FPAD, mT, kk * 32);
                v16h b = load_frag(sW2, FPAD, nT, kk * 32);
                c = __builtin_amdgcn_wmma_f32_16x16x32_f16(false, a, false, b, (short)0, c, false, false);
            }
            int n = nT + (lane & 15), m0 = mT + ((lane >> 4) << 3);
            float bb = b2[n];
#pragma unroll
            for (int v = 0; v < 8; v++)
                wfA[(m0 + v) * FPAD + n] = (_Float16)((c[v] + bb) * ccS[m0 + v]);
        }
        __syncthreads();
        // message scatter-accumulate: thread tid owns feature column tid
        {
            const int f = tid;
#pragma unroll 4
            for (int e = 0; e < ECHUNK; e++) {
                agg[tgtS[e] * FPAD + f] +=
                    (float)x1A[srcS[e] * FPAD + f] * (float)wfA[e * FPAD + f];
            }
        }
        __syncthreads();
    }

    // agg -> f16 (reuse hA)
    for (int idx = tid; idx < APM * FPAD; idx += 128) hA[idx] = (_Float16)agg[idx];
    __syncthreads();

    // g1 = ssp(agg @ lin2 + lin2_b)
    for (int t = wid; t < 16; t += 4) {
        int mT = (t >> 3) << 4, nT = (t & 7) << 4;
        v8f c = {};
#pragma unroll
        for (int kk = 0; kk < 4; kk++) {
            v16h a = load_frag(hA, FPAD, mT, kk * 32);
            v16h b = load_frag(L2p, FPAD, nT, kk * 32);
            c = __builtin_amdgcn_wmma_f32_16x16x32_f16(false, a, false, b, (short)0, c, false, false);
        }
        int n = nT + (lane & 15), m0 = mT + ((lane >> 4) << 3);
        float bb = bl2[n];
#pragma unroll
        for (int v = 0; v < 8; v++) tA[(m0 + v) * FPAD + n] = (_Float16)sspf(c[v] + bb);
    }
    __syncthreads();

    // h += g1 @ lin_w + lin_b
    for (int t = wid; t < 16; t += 4) {
        int mT = (t >> 3) << 4, nT = (t & 7) << 4;
        v8f c = {};
#pragma unroll
        for (int kk = 0; kk < 4; kk++) {
            v16h a = load_frag(tA, FPAD, mT, kk * 32);
            v16h b = load_frag(LWp, FPAD, nT, kk * 32);
            c = __builtin_amdgcn_wmma_f32_16x16x32_f16(false, a, false, b, (short)0, c, false, false);
        }
        int n = nT + (lane & 15), m0 = mT + ((lane >> 4) << 3);
        float bb = blw[n];
#pragma unroll
        for (int v = 0; v < 8; v++)
            hbuf[(size_t)(a0 + m0 + v) * FPAD + n] += c[v] + bb;
    }
}

// ---------------- readout head + per-molecule energy ----------------
__global__ void readout_kernel(const float* hbuf, const float* ow1, const float* ob1,
                               const float* ow2, const float* ob2, float* out) {
    int m = blockIdx.x, a = threadIdx.x;   // 32 threads = 1 wave
    const float* hr = hbuf + (size_t)(m * APM + a) * FPAD;
    float acc = 0.f;
    const int H = FEAT / 2;
    for (int hh = 0; hh < H; hh++) {
        float s = ob1[hh];
        for (int f = 0; f < FEAT; f++) s += hr[f] * ow1[f * H + hh];
        acc += sspf(s) * ow2[hh];
    }
    acc += ob2[0];
    for (int off = 16; off > 0; off >>= 1) acc += __shfl_down(acc, off, 32);
    if (a == 0) out[m] = acc;
}

// ---------------- host-side orchestration ----------------
extern "C" void kernel_launch(void* const* d_in, const int* in_sizes, int n_in,
                              void* d_out, int out_size, void* d_ws, size_t ws_size,
                              hipStream_t stream) {
    const int*   z      = (const int*)d_in[0];
    const float* pos    = (const float*)d_in[1];
    // d_in[2] = ptr (structure is fixed: 32 atoms/molecule)
    const float* emb    = (const float*)d_in[3];
    const float* mw1    = (const float*)d_in[4];
    const float* mb1    = (const float*)d_in[5];
    const float* mw2    = (const float*)d_in[6];
    const float* mb2    = (const float*)d_in[7];
    const float* l1w    = (const float*)d_in[8];
    const float* l2w    = (const float*)d_in[9];
    const float* l2b    = (const float*)d_in[10];
    const float* lww    = (const float*)d_in[11];
    const float* lwb    = (const float*)d_in[12];
    const float* ow1    = (const float*)d_in[13];
    const float* ob1    = (const float*)d_in[14];
    const float* ow2    = (const float*)d_in[15];
    const float* ob2    = (const float*)d_in[16];

    // workspace layout
    char* ws = (char*)d_ws;
    size_t o = 0;
    float*     hbuf = (float*)(ws + o);      o += (size_t)NATOMS * FPAD * 4;     // 4 MB
    float*     ewb  = (float*)(ws + o);      o += (size_t)NEDGE * 4;
    int*       srcb = (int*)(ws + o);        o += (size_t)NEDGE * 4;
    _Float16*  W1p  = (_Float16*)(ws + o);   o += (size_t)NLAYER * FPAD * NGPAD * 2;
    _Float16*  W2p  = (_Float16*)(ws + o);   o += (size_t)NLAYER * FPAD * FPAD * 2;
    _Float16*  L1p  = (_Float16*)(ws + o);   o += (size_t)NLAYER * FPAD * FPAD * 2;
    _Float16*  L2p  = (_Float16*)(ws + o);   o += (size_t)NLAYER * FPAD * FPAD * 2;
    _Float16*  LWp  = (_Float16*)(ws + o);   o += (size_t)NLAYER * FPAD * FPAD * 2;
    float*     B1   = (float*)(ws + o);      o += (size_t)NLAYER * FPAD * 4;
    float*     B2   = (float*)(ws + o);      o += (size_t)NLAYER * FPAD * 4;
    float*     BL2  = (float*)(ws + o);      o += (size_t)NLAYER * FPAD * 4;
    float*     BLW  = (float*)(ws + o);      o += (size_t)NLAYER * FPAD * 4;

    prep_kernel<<<128, 256, 0, stream>>>(mw1, mb1, mw2, mb2, l1w, l2w, l2b, lww, lwb,
                                         W1p, W2p, L1p, L2p, LWp, B1, B2, BL2, BLW);
    embed_kernel<<<(NATOMS * FPAD + 255) / 256, 256, 0, stream>>>(z, emb, hbuf);
    graph_kernel<<<NMOL, APM, 0, stream>>>(pos, ewb, srcb);

    for (int l = 0; l < NLAYER; l++) {
        layer_kernel<<<NMOL, 128, 0, stream>>>(
            hbuf, ewb, srcb,
            W1p + (size_t)l * FPAD * NGPAD,
            W2p + (size_t)l * FPAD * FPAD,
            L1p + (size_t)l * FPAD * FPAD,
            L2p + (size_t)l * FPAD * FPAD,
            LWp + (size_t)l * FPAD * FPAD,
            B1 + l * FPAD, B2 + l * FPAD, BL2 + l * FPAD, BLW + l * FPAD);
    }

    readout_kernel<<<NMOL, APM, 0, stream>>>(hbuf, ow1, ob1, ow2, ob2, (float*)d_out);
}
